// PointNetLearner_53257594470663
// MI455X (gfx1250) — matrix operand order repaired and checked by reading, not compiled, not run on hardware
//
#include <hip/hip_runtime.h>
#include <hip/hip_bf16.h>

#define H 128
#define TILE 256
#define SP 136            // padded LDS row stride (bf16 elems) = 272B -> conflict-free
#define THREADS 256

typedef __attribute__((ext_vector_type(8)))  __bf16 v8bf;
typedef __attribute__((ext_vector_type(16))) __bf16 v16bf;
typedef __attribute__((ext_vector_type(8)))  float  v8f;

// A-fragment (16x32 bf16): lane holds K = [base..base+7] in v0-3, [base+16..base+23] in v4-7
static __device__ __forceinline__ v16bf frag_gap8(const __bf16* p) {
  v8bf lo = *(const v8bf*)(p);
  v8bf hi = *(const v8bf*)(p + 16);
  return __builtin_shufflevector(lo, hi, 0,1,2,3,4,5,6,7,8,9,10,11,12,13,14,15);
}
// B-fragment (32x16 bf16): lane holds K = [base..base+15] contiguous
static __device__ __forceinline__ v16bf frag_contig(const __bf16* p) {
  v8bf lo = *(const v8bf*)(p);
  v8bf hi = *(const v8bf*)(p + 8);
  return __builtin_shufflevector(lo, hi, 0,1,2,3,4,5,6,7,8,9,10,11,12,13,14,15);
}

__global__ void zero_kernel(float* __restrict__ p, int n) {
  int i = blockIdx.x * blockDim.x + threadIdx.x;
  if (i < n) p[i] = 0.f;
}

__global__ void cvt_kernel(const float* __restrict__ src, __bf16* __restrict__ dst, int n) {
  int i = blockIdx.x * blockDim.x + threadIdx.x;
  if (i < n) dst[i] = (__bf16)src[i];
}

// Per-point MLP (layer0 VALU + NL x 128x128 BF16-WMMA layers) + contiguous segment-sum.
// WMMA formulation: D[n][m] = sum_k W[n][k] * x[m][k]  (A = weights, B = activations^T),
// so a D lane owns one point (column) and 8 consecutive output channels (rows) ->
// packed b128 LDS writeback and contiguous atomics.
__global__ __launch_bounds__(THREADS)
void embed_kernel(const float* __restrict__ P, const float* __restrict__ Q,
                  const int* __restrict__ Pidx, const int* __restrict__ Qidx,
                  const float* __restrict__ hW0, const float* __restrict__ hb0,
                  const __bf16* __restrict__ hWb, const float* __restrict__ hb,
                  float* __restrict__ Psum, float* __restrict__ Qsum,
                  int NP, int NQ, int B, int NL)
{
  __shared__ __align__(16) __bf16 xbuf[TILE][SP];   // activations, in-place across layers
  __shared__ __align__(16) __bf16 wbuf[H][SP];      // current layer weights, row = out channel
  __shared__ __align__(32) float  bias[H];
  __shared__ float  w0[H * 3];
  __shared__ float  b0s[H];
  __shared__ int    segid[TILE];

  const int isQ = blockIdx.y;
  const float* pts = isQ ? Q : P;
  const int*   idx = isQ ? Qidx : Pidx;
  float*       Sum = isQ ? Qsum : Psum;
  const int    N   = isQ ? NQ : NP;

  const int tile = blockIdx.x;
  const int ntiles = (N + TILE - 1) / TILE;
  if (tile >= ntiles) return;           // uniform per block: EXEC stays all-ones for WMMA
  const int base = tile * TILE;
  const int t = threadIdx.x;

  // segment id per point via binary search over contiguous range starts
  for (int i = t; i < TILE; i += THREADS) {
    int p = base + i, s = -1;
    if (p < N) {
      int lo = 0, hi = B - 1;
      while (lo < hi) { int mid = (lo + hi + 1) >> 1; if (idx[2 * mid] <= p) lo = mid; else hi = mid - 1; }
      s = lo;
    }
    segid[i] = s;
  }
  for (int i = t; i < H * 3; i += THREADS) w0[i] = hW0[i];
  if (t < H) b0s[t] = hb0[t];
  __syncthreads();

  // ---- layer 0: 3 -> 128 (VALU), packed bf16 activation stores ----
  {
    int p = base + t;
    float c0 = 0.f, c1 = 0.f, c2 = 0.f;
    bool ok = p < N;
    if (ok) { c0 = pts[3 * p]; c1 = pts[3 * p + 1]; c2 = pts[3 * p + 2]; }
    for (int nb = 0; nb < H; nb += 8) {
      v8bf pk;
      #pragma unroll
      for (int j = 0; j < 8; ++j) {
        int n = nb + j;
        float y = fmaxf(fmaf(c0, w0[3 * n], fmaf(c1, w0[3 * n + 1], fmaf(c2, w0[3 * n + 2], b0s[n]))), 0.f);
        pk[j] = ok ? (__bf16)y : (__bf16)0.f;
      }
      *(v8bf*)&xbuf[t][nb] = pk;
    }
  }
  __syncthreads();

  const int lane = t & 31, wave = t >> 5;
  const int lm = lane & 15, half = lane >> 4;
  const int r0a = wave * 32;            // this wave's 32 points
  const int pr0 = r0a + lm;             // point owned in m-tile 0
  const int pr1 = r0a + 16 + lm;        // point owned in m-tile 1

  // ---- NL-1 plain hidden layers ----
  for (int l = 0; l < NL - 1; ++l) {
    // stage weights (vectorized b128 copies) + bias
    const v8bf* Wg8 = (const v8bf*)(hWb + (size_t)l * H * H);
    for (int i = t; i < (H * H) / 8; i += THREADS)
      *(v8bf*)&wbuf[i >> 4][(i & 15) * 8] = Wg8[i];
    if (t < H) bias[t] = hb[l * H + t];
    __builtin_prefetch(hWb + (size_t)(l + 1) * H * H + (size_t)t * 64, 0, 1);
    __syncthreads();

    // activations for both m-tiles, register-resident (in-place update is safe)
    v16bf x00 = frag_contig(&xbuf[pr0][  0 + half * 16]);
    v16bf x01 = frag_contig(&xbuf[pr0][ 32 + half * 16]);
    v16bf x02 = frag_contig(&xbuf[pr0][ 64 + half * 16]);
    v16bf x03 = frag_contig(&xbuf[pr0][ 96 + half * 16]);
    v16bf x10 = frag_contig(&xbuf[pr1][  0 + half * 16]);
    v16bf x11 = frag_contig(&xbuf[pr1][ 32 + half * 16]);
    v16bf x12 = frag_contig(&xbuf[pr1][ 64 + half * 16]);
    v16bf x13 = frag_contig(&xbuf[pr1][ 96 + half * 16]);

    #pragma unroll
    for (int nt = 0; nt < 8; ++nt) {
      const int wrow = nt * 16 + lm;
      v16bf wf0 = frag_contig(&wbuf[wrow][  0 + half * 16]);
      v16bf wf1 = frag_contig(&wbuf[wrow][ 32 + half * 16]);
      v16bf wf2 = frag_contig(&wbuf[wrow][ 64 + half * 16]);
      v16bf wf3 = frag_contig(&wbuf[wrow][ 96 + half * 16]);
      const v8f cb = *(const v8f*)&bias[nt * 16 + half * 8];   // accumulator = bias

      v8f c0 = cb;
      c0 = __builtin_amdgcn_wmma_f32_16x16x32_bf16(false, wf0, false, x00, (short)0, c0, false, false);
      c0 = __builtin_amdgcn_wmma_f32_16x16x32_bf16(false, wf1, false, x01, (short)0, c0, false, false);
      c0 = __builtin_amdgcn_wmma_f32_16x16x32_bf16(false, wf2, false, x02, (short)0, c0, false, false);
      c0 = __builtin_amdgcn_wmma_f32_16x16x32_bf16(false, wf3, false, x03, (short)0, c0, false, false);
      v8f c1 = cb;
      c1 = __builtin_amdgcn_wmma_f32_16x16x32_bf16(false, wf0, false, x10, (short)0, c1, false, false);
      c1 = __builtin_amdgcn_wmma_f32_16x16x32_bf16(false, wf1, false, x11, (short)0, c1, false, false);
      c1 = __builtin_amdgcn_wmma_f32_16x16x32_bf16(false, wf2, false, x12, (short)0, c1, false, false);
      c1 = __builtin_amdgcn_wmma_f32_16x16x32_bf16(false, wf3, false, x13, (short)0, c1, false, false);

      // relu + pack 8 consecutive channels -> single b128 store per tile
      v8bf pk0, pk1;
      #pragma unroll
      for (int i = 0; i < 8; ++i) {
        pk0[i] = (__bf16)fmaxf(c0[i], 0.f);
        pk1[i] = (__bf16)fmaxf(c1[i], 0.f);
      }
      *(v8bf*)&xbuf[pr0][nt * 16 + half * 8] = pk0;
      *(v8bf*)&xbuf[pr1][nt * 16 + half * 8] = pk1;
    }
    __syncthreads();   // protect wbuf before next layer's weight load
  }

  // ---- last hidden layer: relu + F32 segment accumulation (one segid per lane) ----
  {
    const int l = NL - 1;
    const v8bf* Wg8 = (const v8bf*)(hWb + (size_t)l * H * H);
    for (int i = t; i < (H * H) / 8; i += THREADS)
      *(v8bf*)&wbuf[i >> 4][(i & 15) * 8] = Wg8[i];
    if (t < H) bias[t] = hb[l * H + t];
    __syncthreads();

    v16bf x00 = frag_contig(&xbuf[pr0][  0 + half * 16]);
    v16bf x01 = frag_contig(&xbuf[pr0][ 32 + half * 16]);
    v16bf x02 = frag_contig(&xbuf[pr0][ 64 + half * 16]);
    v16bf x03 = frag_contig(&xbuf[pr0][ 96 + half * 16]);
    v16bf x10 = frag_contig(&xbuf[pr1][  0 + half * 16]);
    v16bf x11 = frag_contig(&xbuf[pr1][ 32 + half * 16]);
    v16bf x12 = frag_contig(&xbuf[pr1][ 64 + half * 16]);
    v16bf x13 = frag_contig(&xbuf[pr1][ 96 + half * 16]);
    const int s0 = segid[pr0];
    const int s1 = segid[pr1];

    #pragma unroll
    for (int nt = 0; nt < 8; ++nt) {
      const int wrow = nt * 16 + lm;
      v16bf wf0 = frag_contig(&wbuf[wrow][  0 + half * 16]);
      v16bf wf1 = frag_contig(&wbuf[wrow][ 32 + half * 16]);
      v16bf wf2 = frag_contig(&wbuf[wrow][ 64 + half * 16]);
      v16bf wf3 = frag_contig(&wbuf[wrow][ 96 + half * 16]);
      const v8f cb = *(const v8f*)&bias[nt * 16 + half * 8];

      v8f c0 = cb;
      c0 = __builtin_amdgcn_wmma_f32_16x16x32_bf16(false, wf0, false, x00, (short)0, c0, false, false);
      c0 = __builtin_amdgcn_wmma_f32_16x16x32_bf16(false, wf1, false, x01, (short)0, c0, false, false);
      c0 = __builtin_amdgcn_wmma_f32_16x16x32_bf16(false, wf2, false, x02, (short)0, c0, false, false);
      c0 = __builtin_amdgcn_wmma_f32_16x16x32_bf16(false, wf3, false, x03, (short)0, c0, false, false);
      v8f c1 = cb;
      c1 = __builtin_amdgcn_wmma_f32_16x16x32_bf16(false, wf0, false, x10, (short)0, c1, false, false);
      c1 = __builtin_amdgcn_wmma_f32_16x16x32_bf16(false, wf1, false, x11, (short)0, c1, false, false);
      c1 = __builtin_amdgcn_wmma_f32_16x16x32_bf16(false, wf2, false, x12, (short)0, c1, false, false);
      c1 = __builtin_amdgcn_wmma_f32_16x16x32_bf16(false, wf3, false, x13, (short)0, c1, false, false);

      const int ch = nt * 16 + half * 8;
      if (s0 >= 0) {
        float* dst = Sum + (size_t)s0 * H + ch;
        #pragma unroll
        for (int i = 0; i < 8; ++i) atomicAdd(dst + i, fmaxf(c0[i], 0.f));
      }
      if (s1 >= 0) {
        float* dst = Sum + (size_t)s1 * H + ch;
        #pragma unroll
        for (int i = 0; i < 8; ++i) atomicAdd(dst + i, fmaxf(c1[i], 0.f));
      }
    }
  }
}

// Two 128->128 relu layers on the [2B x 128] segment sums; tiny, VALU is fine.
__global__ __launch_bounds__(H)
void final_kernel(const float* __restrict__ Psum, const float* __restrict__ Qsum,
                  const float* __restrict__ fW, const float* __restrict__ fb,
                  float* __restrict__ Pe, float* __restrict__ Qe, int B)
{
  __shared__ float xr[H];
  const int row = blockIdx.x;
  const float* src = (row < B) ? (Psum + (size_t)row * H) : (Qsum + (size_t)(row - B) * H);
  float*       dst = (row < B) ? (Pe   + (size_t)row * H) : (Qe   + (size_t)(row - B) * H);
  const int t = threadIdx.x;
  xr[t] = src[t];
  __syncthreads();
  float acc = 0.f;
  for (int l = 0; l < 2; ++l) {
    const float* W = fW + (size_t)l * H * H + (size_t)t * H;
    acc = fb[l * H + t];
    #pragma unroll 4
    for (int k = 0; k < H; ++k) acc = fmaf(xr[k], W[k], acc);
    acc = fmaxf(acc, 0.f);
    __syncthreads();
    xr[t] = acc;
    __syncthreads();
  }
  dst[t] = acc * (1.0f / H);
}

// squared-L1 distance, relative-error loss, mean over B -> scalar
__global__ __launch_bounds__(1024)
void loss_kernel(const float* __restrict__ Pe, const float* __restrict__ Qe,
                 const float* __restrict__ dists, float* __restrict__ out, int B)
{
  __shared__ float red[1024];
  const int t = threadIdx.x;
  float v = 0.f;
  for (int b = t; b < B; b += 1024) {
    float s = 0.f;
    #pragma unroll 4
    for (int k = 0; k < H; ++k) s += fabsf(Pe[(size_t)b * H + k] - Qe[(size_t)b * H + k]);
    float d = s * s, dd = dists[b];
    float r = (d - dd) / dd;
    v += r * r + fmaxf(dd - d, 0.f);
  }
  red[t] = v;
  __syncthreads();
  for (int off = 512; off > 0; off >>= 1) {
    if (t < off) red[t] += red[t + off];
    __syncthreads();
  }
  if (t == 0) out[0] = red[0] / (float)B;
}

extern "C" void kernel_launch(void* const* d_in, const int* in_sizes, int n_in,
                              void* d_out, int out_size, void* d_ws, size_t ws_size,
                              hipStream_t stream)
{
  const float* P     = (const float*)d_in[0];
  const float* Q     = (const float*)d_in[1];
  const int*   Pidx  = (const int*)d_in[2];
  const int*   Qidx  = (const int*)d_in[3];
  const float* dists = (const float*)d_in[4];
  const float* hW0   = (const float*)d_in[5];
  const float* hb0   = (const float*)d_in[6];
  const float* hW    = (const float*)d_in[7];
  const float* hb    = (const float*)d_in[8];
  const float* fW    = (const float*)d_in[9];
  const float* fb    = (const float*)d_in[10];

  const int NP = in_sizes[0] / 3;
  const int NQ = in_sizes[1] / 3;
  const int B  = in_sizes[4];
  const int NL = in_sizes[7] / (H * H);   // 9 hidden 128x128 layers

  // workspace layout
  char* ws = (char*)d_ws;
  __bf16* hWb = (__bf16*)ws;
  size_t o = ((size_t)NL * H * H * sizeof(__bf16) + 255) & ~(size_t)255;
  float* Psum = (float*)(ws + o); o += (size_t)B * H * sizeof(float);
  float* Qsum = (float*)(ws + o); o += (size_t)B * H * sizeof(float);
  float* Pe   = (float*)(ws + o); o += (size_t)B * H * sizeof(float);
  float* Qe   = (float*)(ws + o);

  // zero segment accumulators (Psum and Qsum are contiguous)
  const int nz = 2 * B * H;
  zero_kernel<<<(nz + 255) / 256, 256, 0, stream>>>(Psum, nz);

  // convert hidden weights to bf16 once per launch
  const int nc = NL * H * H;
  cvt_kernel<<<(nc + 255) / 256, 256, 0, stream>>>(hW, hWb, nc);

  const int ntP = (NP + TILE - 1) / TILE;
  const int ntQ = (NQ + TILE - 1) / TILE;
  dim3 grid(ntP > ntQ ? ntP : ntQ, 2);
  embed_kernel<<<grid, THREADS, 0, stream>>>(P, Q, Pidx, Qidx, hW0, hb0, hWb, hb,
                                             Psum, Qsum, NP, NQ, B, NL);

  final_kernel<<<2 * B, H, 0, stream>>>(Psum, Qsum, fW, fb, Pe, Qe, B);

  loss_kernel<<<1, 1024, 0, stream>>>(Pe, Qe, dists, (float*)d_out, B);
}
